// GRUEncoder_89034672046893
// MI455X (gfx1250) — compile-verified, matrix operand order checked by reading, not run on hardware
//
#include <hip/hip_runtime.h>

typedef __attribute__((ext_vector_type(16))) _Float16 v16h;
typedef __attribute__((ext_vector_type(8)))  _Float16 v8h;
typedef __attribute__((ext_vector_type(8)))  float    v8f;
typedef __attribute__((ext_vector_type(4)))  unsigned v4u;
typedef __attribute__((ext_vector_type(8)))  int      v8i;
typedef __attribute__((ext_vector_type(4)))  int      v4i;

#define GB 64
#define GT 2048
#define GF 512
#define GH 128

// ---------------------------------------------------------------------------
// Fragment helpers (wave32, V_WMMA_F32_16X16X32_F16 operand layouts):
//   A (16x32 f16, row-major source): lane = 16*half + m holds row m,
//     K = 32*kiter + 8*half + [0..7]  (frag elems 0..7)  and +16 (elems 8..15).
//   B (32x16 f16) from a row-major [N][K] weight (i.e. B = W^T): identical
//     addressing with the lane's low nibble selecting the output column row
//     of W. Both reduce to two contiguous 16-byte loads per lane.
// ---------------------------------------------------------------------------
static __device__ __forceinline__ v16h make_frag(const _Float16* p) {
  v8h lo = *(const v8h*)(p);
  v8h hi = *(const v8h*)(p + 16);
  return __builtin_shufflevector(lo, hi, 0, 1, 2, 3, 4, 5, 6, 7,
                                 8, 9, 10, 11, 12, 13, 14, 15);
}

static __device__ __forceinline__ v16h cvt_frag_f32(const float* p) {
  v16h f;
#pragma unroll
  for (int i = 0; i < 8; ++i) {
    f[i]     = (_Float16)p[i];
    f[i + 8] = (_Float16)p[16 + i];
  }
  return f;
}

static __device__ __forceinline__ v8f wmma16(v16h a, v16h b, v8f c) {
  return __builtin_amdgcn_wmma_f32_16x16x32_f16(false, a, false, b,
                                                (short)0, c, false, false);
}

static __device__ __forceinline__ float sigmoidf_dev(float v) {
  return 1.0f / (1.0f + __expf(-v));
}

static __device__ __forceinline__ float tanhf_dev(float v) {
  const float e = __expf(2.0f * v);          // +inf / 0 at the tails -> +-1
  return 1.0f - 2.0f / (e + 1.0f);
}

// ---------------------------------------------------------------------------
// TDM: async-load one [16 rows x 128 cols] f16 tile (rows strided by GT*GH
// elements) from global `gsrc` into LDS at byte offset `lds_addr`.
// D# per CDNA5 ISA ch.8: group0 = {count=1, lds_addr, global_addr, type=2},
// group1 = {data_size=2B, tensor_dim0=128, tensor_dim1=16, tile=128x16,
// dim0_stride=GT*GH}.  Tracked on TENSORcnt.
// This toolchain exposes the 6-arg builtin: (g0, g1, g2, g3, g4, cpol).
// ---------------------------------------------------------------------------
static __device__ __forceinline__ void tdm_load_feat_tile(
    const _Float16* gsrc, unsigned lds_addr) {
  const unsigned long long ga = (unsigned long long)(uintptr_t)gsrc;
  v4u g0;
  g0[0] = 1u;                                        // count = 1 descriptor
  g0[1] = lds_addr;                                  // LDS byte address
  g0[2] = (unsigned)ga;                              // global addr [31:0]
  g0[3] = ((unsigned)(ga >> 32) & 0x01FFFFFFu)       // global addr [56:32]
          | (2u << 30);                              // type = 2 ("image")
  v8i g1;
  g1[0] = (int)(1u << 16);                           // data_size = 1 (2 bytes)
  g1[1] = (int)(128u << 16);                         // tensor_dim0 = 128
  g1[2] = (int)(16u << 16);                          // tensor_dim1 = 16
  g1[3] = (int)(128u << 16);                         // tile_dim0 = 128
  g1[4] = 16;                                        // tile_dim1 = 16
  g1[5] = GT * GH;                                   // dim0_stride lo (48-bit)
  g1[6] = 0;                                         // dim0_stride hi / d1s lo
  g1[7] = 0;
  v4i gz4 = {0, 0, 0, 0};
  v8i gz8 = {0, 0, 0, 0, 0, 0, 0, 0};
  __builtin_amdgcn_tensor_load_to_lds(g0, g1, gz4, gz4, gz8, 0);
}

// ---------------------------------------------------------------------------
// Kernel 1: ragged lengths.  lengths[b] = #rows t with sum_f x[b,t,f] != 0.
// ---------------------------------------------------------------------------
__global__ __launch_bounds__(256) void compute_lengths(
    const float* __restrict__ x, int* __restrict__ lengths) {
  __shared__ int cnt;
  if (threadIdx.x == 0) cnt = 0;
  __syncthreads();

  const int b    = blockIdx.x;
  const int wave = threadIdx.x >> 5;
  const int lane = threadIdx.x & 31;

  int local = 0;
  for (int t = wave; t < GT; t += 8) {
    const float* row = x + ((size_t)b * GT + t) * GF;
    float s = 0.0f;
    for (int j = lane; j < GF; j += 32) s += row[j];
#pragma unroll
    for (int off = 16; off > 0; off >>= 1) s += __shfl_xor(s, off, 32);
    if (lane == 0 && s != 0.0f) ++local;
  }
  if (lane == 0) atomicAdd(&cnt, local);
  __syncthreads();
  if (threadIdx.x == 0) lengths[b] = cnt;
}

// ---------------------------------------------------------------------------
// Kernel 2: one-time fp32 -> f16 weight conversion (tiny; L2-resident after).
// ---------------------------------------------------------------------------
__global__ void cvt_weights(const float* __restrict__ wproj,
                            const float* __restrict__ wih,
                            const float* __restrict__ whh,
                            _Float16* __restrict__ wproj_h,
                            _Float16* __restrict__ wih_h,
                            _Float16* __restrict__ whh_h) {
  const int stride = gridDim.x * blockDim.x;
  for (int i = blockIdx.x * blockDim.x + threadIdx.x; i < GH * GF; i += stride)
    wproj_h[i] = (_Float16)wproj[i];
  for (int i = blockIdx.x * blockDim.x + threadIdx.x; i < 2 * 3 * GH * GH;
       i += stride) {
    wih_h[i] = (_Float16)wih[i];
    whh_h[i] = (_Float16)whh[i];
  }
}

// ---------------------------------------------------------------------------
// Kernel 3: projection GEMM  feat[B*T, H] = x[B*T, F] @ w_proj^T + b_proj,
// emitted as f16.  Bandwidth-bound (256 MB of x).  All 8 B-fragments are
// loaded before the 8-WMMA chain so the b128 clause keeps loads in flight.
// ---------------------------------------------------------------------------
__global__ __launch_bounds__(256, 2) void proj_gemm(
    const float* __restrict__ x, const _Float16* __restrict__ wproj_h,
    const float* __restrict__ b_proj, _Float16* __restrict__ feat) {
  const int tid  = threadIdx.x;
  const int wave = tid >> 5;
  const int lane = tid & 31;
  const int m    = lane & 15;
  const int half = lane >> 4;
  const size_t row0 = ((size_t)blockIdx.x * 8 + wave) * 16;

  v8f acc[8];
#pragma unroll
  for (int n = 0; n < 8; ++n) {
    const float bv = b_proj[n * 16 + m];
#pragma unroll
    for (int e = 0; e < 8; ++e) acc[n][e] = bv;
  }

  for (int k = 0; k < GF / 32; ++k) {
    const float* pA = x + (row0 + m) * GF + 32 * k + 8 * half;
    v16h bfr[8];
#pragma unroll
    for (int n = 0; n < 8; ++n) {
      const _Float16* pB =
          wproj_h + (size_t)(n * 16 + m) * GF + 32 * k + 8 * half;
      bfr[n] = make_frag(pB);
    }
    v16h a = cvt_frag_f32(pA);
#pragma unroll
    for (int n = 0; n < 8; ++n) acc[n] = wmma16(a, bfr[n], acc[n]);
  }

#pragma unroll
  for (int n = 0; n < 8; ++n) {
#pragma unroll
    for (int e = 0; e < 8; ++e) {
      const size_t row = row0 + e + 8 * half;
      feat[row * GH + n * 16 + m] = (_Float16)acc[n][e];
    }
  }
}

// ---------------------------------------------------------------------------
// Kernel 4: sequential GRU scan.  4 blocks (16 batch rows each), 768 threads
// = 24 waves; wave w owns gate-column tile [16w, 16w+16).  Waves 0..15: r/z
// (one fused WMMA chain); waves 16..23: n (separate gx/gh).  Weight
// B-fragments are register-resident for the whole scan.  The next timestep's
// feature tile is TDM-prefetched into a double-buffered LDS tile while the
// current step computes (s_wait_tensorcnt before the step's last barrier).
// ---------------------------------------------------------------------------
__global__ __launch_bounds__(768, 1) void gru_scan(
    const _Float16* __restrict__ feat, const _Float16* __restrict__ wih_h,
    const _Float16* __restrict__ whh_h, const float* __restrict__ b_ih,
    const float* __restrict__ b_hh, const int* __restrict__ lengths,
    float* __restrict__ out) {
  __shared__ __align__(16) float    hstate[2][16][GH];   // fp32 master state
  __shared__ __align__(16) _Float16 h16[2][16][GH];      // f16 copy (A operand)
  __shared__ __align__(16) _Float16 featbuf[2][16][GH];  // TDM double buffer
  __shared__ __align__(16) float    grz[16][256];        // combined r|z gates
  __shared__ __align__(16) _Float16 gnx[16][GH];         // xn part
  __shared__ __align__(16) _Float16 gnh[16][GH];         // hn part
  __shared__ int len_s[16];

  const int tid   = threadIdx.x;
  const int wave  = tid >> 5;
  const int lane  = tid & 31;
  const int m     = lane & 15;
  const int half  = lane >> 4;
  const int bbase = blockIdx.x * 16;

  // Prologue: kick off the TDM load of the t=0 feature tile, then init state.
  if (wave == 0) {
    tdm_load_feat_tile(feat + (size_t)bbase * GT * GH,
                       (unsigned)(uintptr_t)&featbuf[0][0][0]);
  }
  for (int i = tid; i < 2 * 16 * GH; i += 768) {
    (&hstate[0][0][0])[i] = 0.0f;
    (&h16[0][0][0])[i]    = (_Float16)0.0f;
  }
  if (tid < 16) len_s[tid] = lengths[bbase + tid];
  if (wave == 0) __builtin_amdgcn_s_wait_tensorcnt(0);
  __syncthreads();

  const bool is_rz = (wave < 16);
  const int  nt    = wave;           // N tile 0..23 over the 384 gate cols
  const int  ncol  = nt * 16 + m;    // this lane's gate column

  // Register-resident weight fragments: 2 layers x {W_ih, W_hh} x 4 K-slices.
  v16h wih_f[2][4], whh_f[2][4];
#pragma unroll
  for (int l = 0; l < 2; ++l) {
#pragma unroll
    for (int k = 0; k < 4; ++k) {
      const size_t off = (size_t)l * 3 * GH * GH + (size_t)ncol * GH +
                         32 * k + 8 * half;
      wih_f[l][k] = make_frag(wih_h + off);
      whh_f[l][k] = make_frag(whh_h + off);
    }
  }

  float bi[2], bh[2];
#pragma unroll
  for (int l = 0; l < 2; ++l) {
    bi[l] = b_ih[l * 3 * GH + ncol];
    bh[l] = b_hh[l * 3 * GH + ncol];
  }

  for (int t = 0; t < GT; ++t) {
    const int buf = t & 1;
    // Async prefetch of the next timestep's tile into the other buffer.
    if (wave == 0 && t + 1 < GT) {
      tdm_load_feat_tile(feat + ((size_t)bbase * GT + (t + 1)) * GH,
                         (unsigned)(uintptr_t)&featbuf[buf ^ 1][0][0]);
    }
#pragma unroll
    for (int l = 0; l < 2; ++l) {
      if (is_rz) {
        v8f acc;
#pragma unroll
        for (int e = 0; e < 8; ++e) acc[e] = bi[l] + bh[l];
#pragma unroll
        for (int k = 0; k < 4; ++k) {
          v16h a = (l == 0)
                       ? make_frag(&featbuf[buf][m][32 * k + 8 * half])
                       : make_frag(&h16[0][m][32 * k + 8 * half]);
          acc = wmma16(a, wih_f[l][k], acc);
        }
#pragma unroll
        for (int k = 0; k < 4; ++k) {
          v16h a = make_frag(&h16[l][m][32 * k + 8 * half]);
          acc = wmma16(a, whh_f[l][k], acc);
        }
#pragma unroll
        for (int e = 0; e < 8; ++e) grz[e + 8 * half][ncol] = acc[e];
      } else {
        v8f accx, acch;
#pragma unroll
        for (int e = 0; e < 8; ++e) { accx[e] = bi[l]; acch[e] = bh[l]; }
#pragma unroll
        for (int k = 0; k < 4; ++k) {
          v16h a = (l == 0)
                       ? make_frag(&featbuf[buf][m][32 * k + 8 * half])
                       : make_frag(&h16[0][m][32 * k + 8 * half]);
          accx = wmma16(a, wih_f[l][k], accx);
        }
#pragma unroll
        for (int k = 0; k < 4; ++k) {
          v16h a = make_frag(&h16[l][m][32 * k + 8 * half]);
          acch = wmma16(a, whh_f[l][k], acch);
        }
        const int col = ncol - 256;
#pragma unroll
        for (int e = 0; e < 8; ++e) {
          gnx[e + 8 * half][col] = (_Float16)accx[e];
          gnh[e + 8 * half][col] = (_Float16)acch[e];
        }
      }
      __syncthreads();

      // Fused gates: r,z = sigmoid; n = tanh(xn + r*hn); blend + length freeze.
      for (int i = tid; i < 16 * GH; i += 768) {
        const int mm = i >> 7;
        const int j  = i & (GH - 1);
        const float r  = sigmoidf_dev(grz[mm][j]);
        const float z  = sigmoidf_dev(grz[mm][j + GH]);
        const float n  = tanhf_dev((float)gnx[mm][j] + r * (float)gnh[mm][j]);
        const float hp = hstate[l][mm][j];
        float hn = (1.0f - z) * n + z * hp;
        hn = (t < len_s[mm]) ? hn : hp;
        hstate[l][mm][j] = hn;
        h16[l][mm][j]    = (_Float16)hn;
      }
      // Retire the TDM prefetch before the barrier that publishes this step.
      if (l == 1 && wave == 0) __builtin_amdgcn_s_wait_tensorcnt(0);
      __syncthreads();
    }
  }

  for (int i = tid; i < 16 * GH; i += 768) {
    const int mm = i >> 7;
    const int j  = i & (GH - 1);
    out[(size_t)(bbase + mm) * GH + j] = hstate[1][mm][j];
  }
}

// ---------------------------------------------------------------------------
extern "C" void kernel_launch(void* const* d_in, const int* in_sizes, int n_in,
                              void* d_out, int out_size, void* d_ws,
                              size_t ws_size, hipStream_t stream) {
  (void)in_sizes; (void)n_in; (void)out_size; (void)ws_size;

  const float* x      = (const float*)d_in[0];
  const float* w_proj = (const float*)d_in[1];
  const float* b_proj = (const float*)d_in[2];
  const float* w_ih   = (const float*)d_in[3];
  const float* w_hh   = (const float*)d_in[4];
  const float* b_ih   = (const float*)d_in[5];
  const float* b_hh   = (const float*)d_in[6];
  float* out = (float*)d_out;

  // Workspace layout (~34.1 MB):
  _Float16* feat_h  = (_Float16*)d_ws;                   // B*T*H f16 = 32 MB
  _Float16* wih_h   = feat_h + (size_t)GB * GT * GH;     // 2*384*128 f16
  _Float16* whh_h   = wih_h + (size_t)2 * 3 * GH * GH;   // 2*384*128 f16
  _Float16* wproj_h = whh_h + (size_t)2 * 3 * GH * GH;   // 128*512 f16
  int* lengths      = (int*)(wproj_h + (size_t)GH * GF); // 64 ints

  compute_lengths<<<GB, 256, 0, stream>>>(x, lengths);
  cvt_weights<<<256, 256, 0, stream>>>(w_proj, w_ih, w_hh,
                                       wproj_h, wih_h, whh_h);
  proj_gemm<<<(GB * GT) / 128, 256, 0, stream>>>(x, wproj_h, b_proj, feat_h);
  gru_scan<<<GB / 16, 768, 0, stream>>>(feat_h, wih_h, whh_h,
                                        b_ih, b_hh, lengths, out);
}
